// OTLoss_62998580298234
// MI455X (gfx1250) — compile-verified, matrix operand order checked by reading, not run on hardware
//
#include <hip/hip_runtime.h>
#include <hip/hip_bf16.h>
#include <math.h>

typedef __attribute__((ext_vector_type(16))) __bf16 v16bf;
typedef __attribute__((ext_vector_type(8)))  float  v8f;

#define NROW 4096
#define KDIM 1024
#define NN   ((size_t)NROW * (size_t)NROW)
#define OT_GAMMA 1.0f
#define OT_EPS_  1e-6f
#define L2_EPS_  1e-12f

union Frag { v16bf v; uint4 q[2]; };

// ---------------------------------------------------------------- normalize
// One block per row of X = [ft; fs]; write bf16 normalized rows to ws.
__global__ void k_normalize(const float* __restrict__ ft, const float* __restrict__ fs,
                            __bf16* __restrict__ Xt, __bf16* __restrict__ Xs) {
    int row = blockIdx.x;
    const float* src;
    __bf16* dst;
    if (row < NROW) { src = ft + (size_t)row * KDIM;          dst = Xt + (size_t)row * KDIM; }
    else            { src = fs + (size_t)(row - NROW) * KDIM; dst = Xs + (size_t)(row - NROW) * KDIM; }
    int tid = threadIdx.x;
    float4 xv = ((const float4*)src)[tid];            // 256 threads * 4 = 1024
    float ss = xv.x * xv.x + xv.y * xv.y + xv.z * xv.z + xv.w * xv.w;
    __shared__ float red[256];
    red[tid] = ss; __syncthreads();
    for (int s = 128; s > 0; s >>= 1) { if (tid < s) red[tid] += red[tid + s]; __syncthreads(); }
    float inv = 1.0f / fmaxf(sqrtf(red[0]), L2_EPS_);
    union { __bf16 h[4]; uint2 u2; } o;
    o.h[0] = (__bf16)(xv.x * inv); o.h[1] = (__bf16)(xv.y * inv);
    o.h[2] = (__bf16)(xv.z * inv); o.h[3] = (__bf16)(xv.w * inv);
    ((uint2*)dst)[tid] = o.u2;
}

// ---------------------------------------------------------------- GEMM
// M_raw = Xt (4096x1024) @ Xs^T (1024x4096), bf16 inputs, f32 accum.
// Wave tile: 32x64 (2 A-frags x 4 B-frags -> 8 WMMA 16x16x32 per K-step).
// Block: 8 waves (2 row-groups x 4 col-groups) -> 64x256 per block.
__global__ void __launch_bounds__(256) k_gemm(const __bf16* __restrict__ Xt,
                                              const __bf16* __restrict__ Xs,
                                              float* __restrict__ Mout) {
    int tid  = threadIdx.x;
    int lane = tid & 31, wave = tid >> 5;
    int wr = wave & 1, wc = wave >> 1;
    int row0 = blockIdx.y * 64  + wr * 32;
    int col0 = blockIdx.x * 256 + wc * 64;
    int r15   = lane & 15;
    int khalf = (lane >> 4) << 3;   // lanes 16-31 start at K+8 per ISA 16-bit layout

    const __bf16* at[2];
    at[0] = Xt + (size_t)(row0 + r15) * KDIM + khalf;
    at[1] = at[0] + 16 * KDIM;
    const __bf16* bs[4];
#pragma unroll
    for (int c = 0; c < 4; ++c)
        bs[c] = Xs + (size_t)(col0 + c * 16 + r15) * KDIM + khalf;

    v8f acc[2][4];
#pragma unroll
    for (int r = 0; r < 2; ++r)
#pragma unroll
        for (int c = 0; c < 4; ++c) acc[r][c] = (v8f){};

    for (int k0 = 0; k0 < KDIM; k0 += 32) {
        Frag a[2], b[4];
#pragma unroll
        for (int r = 0; r < 2; ++r) {
            a[r].q[0] = *(const uint4*)(at[r] + k0);
            a[r].q[1] = *(const uint4*)(at[r] + k0 + 16);
        }
#pragma unroll
        for (int c = 0; c < 4; ++c) {
            b[c].q[0] = *(const uint4*)(bs[c] + k0);
            b[c].q[1] = *(const uint4*)(bs[c] + k0 + 16);
        }
#pragma unroll
        for (int r = 0; r < 2; ++r)
#pragma unroll
            for (int c = 0; c < 4; ++c)
                acc[r][c] = __builtin_amdgcn_wmma_f32_16x16x32_bf16(
                    false, a[r].v, false, b[c].v, (short)0, acc[r][c], false, false);
    }

    int hi = lane >> 4;
#pragma unroll
    for (int r = 0; r < 2; ++r) {
        float* orow = Mout + (size_t)(row0 + r * 16 + hi * 8) * NROW + col0 + r15;
#pragma unroll
        for (int c = 0; c < 4; ++c) {
#pragma unroll
            for (int q = 0; q < 8; ++q)
                orow[(size_t)q * NROW + c * 16] = acc[r][c][q];
        }
    }
}

// ---------------------------------------------------------------- stats (mean/std, ddof=1)
__global__ void k_stats(const float* __restrict__ Mraw,
                        double* __restrict__ partS, double* __restrict__ partSS) {
    int b = blockIdx.x, tid = threadIdx.x;
    const float* p = Mraw + (size_t)b * NROW;
    float s = 0.f, ss = 0.f;
    for (int j = tid; j < NROW; j += 256) { float x = p[j]; s += x; ss += x * x; }
    __shared__ double rs[256], rss[256];
    rs[tid] = (double)s; rss[tid] = (double)ss; __syncthreads();
    for (int t = 128; t > 0; t >>= 1) {
        if (tid < t) { rs[tid] += rs[tid + t]; rss[tid] += rss[tid + t]; } __syncthreads();
    }
    if (tid == 0) { partS[b] = rs[0]; partSS[b] = rss[0]; }
}

__global__ void k_statsfin(const double* __restrict__ partS, const double* __restrict__ partSS,
                           float* __restrict__ fscal) {
    int tid = threadIdx.x;
    double s = 0.0, ss = 0.0;
    for (int b = tid; b < NROW; b += 256) { s += partS[b]; ss += partSS[b]; }
    __shared__ double rs[256], rss[256];
    rs[tid] = s; rss[tid] = ss; __syncthreads();
    for (int t = 128; t > 0; t >>= 1) {
        if (tid < t) { rs[tid] += rs[tid + t]; rss[tid] += rss[tid + t]; } __syncthreads();
    }
    if (tid == 0) {
        double N = (double)NN;
        double mean = rs[0] / N;
        double var  = (rss[0] - N * mean * mean) / (N - 1.0);
        fscal[0] = (float)mean;
        fscal[1] = (float)(1.0 / sqrt(var));
    }
}

// ---------------------------------------------------------------- standardize + row max
__global__ void k_standardize(float* __restrict__ Mm, const float* __restrict__ fscal,
                              float* __restrict__ rowmax) {
    int i = blockIdx.x, tid = threadIdx.x;
    float mean = fscal[0], invstd = fscal[1];
    float* p = Mm + (size_t)i * NROW;
    float mx = -3.4e38f;
    for (int j = tid; j < NROW; j += 256) {
        float m = (p[j] - mean) * invstd;
        p[j] = m;
        mx = fmaxf(mx, m);
    }
    __shared__ float red[256];
    red[tid] = mx; __syncthreads();
    for (int t = 128; t > 0; t >>= 1) { if (tid < t) red[tid] = fmaxf(red[tid], red[tid + t]); __syncthreads(); }
    if (tid == 0) rowmax[i] = red[0];
}

// ---------------------------------------------------------------- Sinkhorn (u,v form)
// P = diag(u) * exp(gamma*(M - rowmax)) * diag(v)
__global__ void k_rowpass(const float* __restrict__ Mm, const float* __restrict__ rowmax,
                          const float* __restrict__ v, float* __restrict__ u,
                          const int* __restrict__ flags, float numer, int guarded) {
    if (guarded && flags[0]) return;
    int i = blockIdx.x, tid = threadIdx.x;
    const float* p = Mm + (size_t)i * NROW;
    float rm = rowmax[i];
    float t = 0.f;
    for (int j = tid; j < NROW; j += 256)
        t += expf(OT_GAMMA * (p[j] - rm)) * v[j];
    __shared__ float red[256];
    red[tid] = t; __syncthreads();
    for (int s = 128; s > 0; s >>= 1) { if (tid < s) red[tid] += red[tid + s]; __syncthreads(); }
    if (tid == 0) u[i] = numer / red[0];
}

__global__ void k_prep(unsigned int* __restrict__ umax) { *umax = 0u; }

__global__ void k_colpass(const float* __restrict__ Mm, const float* __restrict__ rowmax,
                          const float* __restrict__ u, float* __restrict__ v,
                          float* __restrict__ svec, unsigned int* __restrict__ umax,
                          const int* __restrict__ flags, int finalMode) {
    if (!finalMode && flags[0]) return;
    int tid = threadIdx.x;
    int j = blockIdx.x * 256 + tid;
    float s = 0.f;
    for (int i = 0; i < NROW; ++i)
        s += u[i] * expf(OT_GAMMA * (Mm[(size_t)i * NROW + j] - rowmax[i]));
    if (finalMode) { v[j] = 1.0f / s; return; }
    svec[j] = s;
    float dev = fabsf(v[j] * s - (1.0f / NROW));
    __shared__ float red[256];
    red[tid] = dev; __syncthreads();
    for (int t = 128; t > 0; t >>= 1) { if (tid < t) red[tid] = fmaxf(red[tid], red[tid + t]); __syncthreads(); }
    if (tid == 0) atomicMax(umax, __float_as_uint(red[0]));
}

__global__ void k_finish(float* __restrict__ v, const float* __restrict__ svec,
                         const unsigned int* __restrict__ umax, int* __restrict__ flags) {
    if (flags[0]) return;
    float maxdev = __uint_as_float(*umax);
    if (maxdev <= OT_EPS_) { if (threadIdx.x == 0) flags[0] = 1; return; }
    for (int j = threadIdx.x; j < NROW; j += 256) v[j] = (1.0f / NROW) / svec[j];
}

// ---------------------------------------------------------------- materialize P + loss pieces
__global__ void k_materialize(const float* __restrict__ Mm, const float* __restrict__ rowmax,
                              const float* __restrict__ u, const float* __restrict__ v,
                              float* __restrict__ Pout,
                              double* __restrict__ partP2, double* __restrict__ partTr) {
    int i = blockIdx.x, tid = threadIdx.x;
    float ui = u[i], rm = rowmax[i];
    const float* mrow = Mm + (size_t)i * NROW;
    float* prow = Pout + (size_t)i * NROW;
    double p2 = 0.0, tr = 0.0;
    for (int j = tid; j < NROW; j += 256) {
        float p = ui * expf(OT_GAMMA * (mrow[j] - rm)) * v[j];
        prow[j] = p;
        p2 += (double)p * (double)p;
        if (j == i) tr += (double)p;
    }
    __shared__ double r2[256], rt[256];
    r2[tid] = p2; rt[tid] = tr; __syncthreads();
    for (int t = 128; t > 0; t >>= 1) {
        if (tid < t) { r2[tid] += r2[tid + t]; rt[tid] += rt[tid + t]; } __syncthreads();
    }
    if (tid == 0) { partP2[i] = r2[0]; partTr[i] = rt[0]; }
}

__global__ void k_loss(const double* __restrict__ partP2, const double* __restrict__ partTr,
                       float* __restrict__ out) {
    int tid = threadIdx.x;
    double s2 = 0.0, st = 0.0;
    for (int b = tid; b < NROW; b += 256) { s2 += partP2[b]; st += partTr[b]; }
    __shared__ double r2[256], rt[256];
    r2[tid] = s2; rt[tid] = st; __syncthreads();
    for (int t = 128; t > 0; t >>= 1) {
        if (tid < t) { r2[tid] += r2[tid + t]; rt[tid] += rt[tid + t]; } __syncthreads();
    }
    if (tid == 0) out[0] = (float)sqrt(r2[0] - 2.0 * rt[0] + (double)NROW);
}

// ---------------------------------------------------------------- init ws state each call
__global__ void k_init(float* __restrict__ u, float* __restrict__ v,
                       int* __restrict__ flags, unsigned int* __restrict__ umax) {
    int tid = threadIdx.x;
    for (int j = tid; j < NROW; j += 256) { u[j] = 1.0f; v[j] = 1.0f; }
    if (tid == 0) { flags[0] = 0; *umax = 0u; }
}

// ---------------------------------------------------------------- launch
extern "C" void kernel_launch(void* const* d_in, const int* in_sizes, int n_in,
                              void* d_out, int out_size, void* d_ws, size_t ws_size,
                              hipStream_t stream) {
    (void)in_sizes; (void)n_in; (void)out_size; (void)ws_size;
    const float* ft = (const float*)d_in[0];
    const float* fs = (const float*)d_in[1];

    float* out  = (float*)d_out;
    float* Pout = out + 1;            // 16M floats
    float* Mm   = out + 1 + NN;       // 16M floats (M region; raw then standardized in place)

    // ---- workspace layout: bf16 operands first, then doubles, then f32 vectors
    char* w = (char*)d_ws;
    __bf16* Xt = (__bf16*)w;                           //  8 MiB
    __bf16* Xs = Xt + (size_t)NROW * KDIM;             //  8 MiB
    double* partS  = (double*)(Xs + (size_t)NROW * KDIM);  // 32 KiB (8-aligned @16MiB)
    double* partSS = partS  + NROW;
    double* partP2 = partSS + NROW;
    double* partTr = partP2 + NROW;
    float* rowmax = (float*)(partTr + NROW);
    float* u      = rowmax + NROW;
    float* v      = u + NROW;
    float* svec   = v + NROW;
    float* fscal  = svec + NROW;                       // [0]=mean [1]=invstd [2]=maxdev bits
    int*   flags  = (int*)(fscal + 4);
    unsigned int* umax = (unsigned int*)&fscal[2];

    // 0) init per-call state
    k_init<<<1, 256, 0, stream>>>(u, v, flags, umax);

    // 1) row-normalize ft/fs -> bf16
    k_normalize<<<2 * NROW, 256, 0, stream>>>(ft, fs, Xt, Xs);

    // 2) M_raw = Xt @ Xs^T (bf16 WMMA, f32 accum), into d_out M region
    k_gemm<<<dim3(NROW / 256, NROW / 64, 1), 256, 0, stream>>>(Xt, Xs, Mm);

    // 3) mean / std (ddof=1)
    k_stats<<<NROW, 256, 0, stream>>>(Mm, partS, partSS);
    k_statsfin<<<1, 256, 0, stream>>>(partS, partSS, fscal);

    // 4) standardize M in place + per-row max
    k_standardize<<<NROW, 256, 0, stream>>>(Mm, fscal, rowmax);

    // 5) Sinkhorn, 20 iterations, u/v scaling form with early-done freeze
    const float rconst = 1.0f / (float)NROW;
    for (int it = 0; it < 20; ++it) {
        k_rowpass<<<NROW, 256, 0, stream>>>(Mm, rowmax, v, u, flags, rconst, 1);
        k_prep<<<1, 1, 0, stream>>>(umax);
        k_colpass<<<NROW / 256, 256, 0, stream>>>(Mm, rowmax, u, v, svec, umax, flags, 0);
        k_finish<<<1, 256, 0, stream>>>(v, svec, umax, flags);
    }

    // 6) post-loop row then column normalization (unguarded)
    k_rowpass<<<NROW, 256, 0, stream>>>(Mm, rowmax, v, u, flags, 1.0f, 0);
    k_colpass<<<NROW / 256, 256, 0, stream>>>(Mm, rowmax, u, v, svec, umax, flags, 1);

    // 7) materialize P, reduce ||P||^2 and trace(P), loss
    k_materialize<<<NROW, 256, 0, stream>>>(Mm, rowmax, u, v, Pout, partP2, partTr);
    k_loss<<<1, 256, 0, stream>>>(partP2, partTr, out);
}
